// ScatterFormerBlock_80049600463071
// MI455X (gfx1250) — matrix operand order repaired and checked by reading, not compiled, not run on hardware
//
#include <hip/hip_runtime.h>
#include <hip/hip_bf16.h>
#include <math.h>

// ---------------- problem constants (from reference) ----------------
#define NPTS   40000
#define DIM    128
#define NH     8
#define DH     16
#define FF     256
#define WIN    12
#define GRID_  468
#define BATCH  2
#define NWX    (GRID_ / WIN)          // 39
#define NWIN   (BATCH * NWX * NWX)    // 3042
#define GC     (DIM / 4)              // 32
#define KS     (WIN + 1)              // 13
#define PAD_   (KS / 2)               // 6
#define MAPSZ  (BATCH * GRID_ * GRID_)
#define EPS_   1e-5f

typedef __bf16 bf16;
typedef __attribute__((ext_vector_type(16))) __bf16 v16bf;
typedef __attribute__((ext_vector_type(8)))  __bf16 v8bf;
typedef __attribute__((ext_vector_type(8)))  float  v8f;

// ---------------- workspace layout (bytes) ----------------
#define MB (size_t)(1024*1024)
#define OFF_BF0  ((size_t)0)         // N*128 bf16  (xbf / attnbf / src1bf / hbf)
#define OFF_BF1  (11*MB)             // N*256 bf16  (h1 / t1)
#define OFF_QKV  (32*MB)             // N*384 f32 ; later reused for A (NWIN*8*256 f32)
#define OFF_Y    (94*MB)             // N*128 f32  (proj out / h2 / t2)
#define OFF_QN   (115*MB)            // N*128 bf16
#define OFF_KN   (126*MB)            // N*128 bf16
#define OFF_V    (137*MB)            // N*128 f32
#define OFF_S1   (158*MB)            // N*128 f32
#define OFF_S2   (179*MB)            // N*128 f32
#define OFF_MAP  (200*MB)            // MAPSZ int32
#define OFF_PW   (202*MB)            // packed bf16 weights
// packed weight sub-offsets (bf16 elements)
#define PW_QKV 0
#define PW_PROJ (PW_QKV + 128*384)
#define PW_W1   (PW_PROJ + 128*128)
#define PW_W2   (PW_W1   + 128*256)
#define PW_CW1  (PW_W2   + 256*128)
#define PW_CW2  (PW_CW1  + 128*256)

// ---------------- weight repack: f32 [K,M] -> WMMA B-operand tiles ----------------
// Tile (kt,nt) is a 32x16 block; lane = n%16 + 16*(k%32>=16); half = k%16.
__global__ void k_pack_w(const float* __restrict__ W, bf16* __restrict__ P, int K, int M) {
  int i = blockIdx.x * blockDim.x + threadIdx.x;
  if (i >= K * M) return;
  int k = i / M, n = i % M;
  int kt = k >> 5, kk = k & 31, nt = n >> 4, nn = n & 15;
  int lane = nn + ((kk >= 16) ? 16 : 0);
  int half = kk & 15;
  int tilesN = M >> 4;
  P[((size_t)(kt * tilesN + nt)) * 512 + lane * 16 + half] = (bf16)W[i];
}

// ---------------- generic WMMA bf16 GEMM ----------------
// Wave owns one 16-row strip and a group of up to 8 column tiles: A fragments
// for the full K depth are loaded ONCE into registers (KT*8 VGPRs/lane), then
// reused across all column tiles -> A is read from memory exactly once per
// group of 8 output tiles instead of once per tile.
// A: row-major bf16 [nrows,K]; Bp: packed tiles; EPI=0: f32 out + bias;
// EPI=1: bf16 out + bias + relu.  KT = K/32 (compile-time).
template <int EPI, int KT>
__global__ void k_gemm_wmma(const bf16* __restrict__ A, const bf16* __restrict__ Bp,
                            const float* __restrict__ bias,
                            float* __restrict__ outF, bf16* __restrict__ outB,
                            int nrows, int M) {
  const int K = KT * 32;
  int lane = threadIdx.x & 31;
  int tilesN = M >> 4;
  int groups = (tilesN + 7) >> 3;                 // 8 column tiles per wave
  int total = (nrows >> 4) * groups;
  int t = blockIdx.x * (blockDim.x >> 5) + (threadIdx.x >> 5);
  if (t >= total) return;                         // whole wave exits together
  int rt = t / groups, gi = t % groups;
  int ct0 = gi * 8;
  int ct1 = (ct0 + 8 < tilesN) ? (ct0 + 8) : tilesN;

  int m = lane & 15, g = lane >> 4;
  const bf16* arow = A + (size_t)(rt * 16 + m) * K;

  // Load all A fragments for this row strip (per-lane: two contiguous 16B
  // chunks per 32-deep K tile, k = 16*g + s, same k order as the packed B tiles).
  union AV { v16bf v; v8bf h[2]; } a[KT];
#pragma unroll
  for (int kt = 0; kt < KT; ++kt) {
    const bf16* pa = arow + kt * 32 + g * 16;
    a[kt].h[0] = *(const v8bf*)(pa);
    a[kt].h[1] = *(const v8bf*)(pa + 8);
  }

  const bf16* bbase = Bp + (size_t)lane * 16;
  int rowbase = rt * 16 + g * 8;

  for (int ct = ct0; ct < ct1; ++ct) {
    v8f c = {};
#pragma unroll
    for (int kt = 0; kt < KT; ++kt) {
      v16bf b = *(const v16bf*)(bbase + ((size_t)(kt * tilesN + ct)) * 512);
      c = __builtin_amdgcn_wmma_f32_16x16x32_bf16(false, a[kt].v, false, b,
                                                  (short)0, c, false, false);
    }
    if (ct + 1 < ct1)
      __builtin_prefetch(bbase + ((size_t)(ct + 1)) * 512, 0, 3);
    int col = ct * 16 + m;
    float bv = bias ? bias[col] : 0.f;
#pragma unroll
    for (int r = 0; r < 8; ++r) {
      float v = c[r] + bv;
      size_t o = (size_t)(rowbase + r) * M + col;
      if (EPI == 0) outF[o] = v;
      else          outB[o] = (bf16)fmaxf(v, 0.f);
    }
  }
}

// ---------------- x = src + pe[win_pos], bf16 ----------------
__global__ void k_embed(const float* __restrict__ src, const int* __restrict__ win_pos,
                        const float* __restrict__ pe, bf16* __restrict__ xbf) {
  int i = blockIdx.x * blockDim.x + threadIdx.x;
  if (i >= NPTS * DIM) return;
  int n = i >> 7, d = i & 127;
  int p = win_pos[n * 2] * WIN + win_pos[n * 2 + 1];
  xbf[i] = (bf16)(src[i] + pe[p * DIM + d]);
}

// ---------------- split qkv, per-head L2-normalize q,k ----------------
__global__ void k_qknorm(const float* __restrict__ qkv, bf16* __restrict__ qn,
                         bf16* __restrict__ kn, float* __restrict__ v) {
  int i = blockIdx.x * blockDim.x + threadIdx.x;   // n*8 + h
  if (i >= NPTS * NH) return;
  int n = i >> 3, h = i & 7;
  const float* row = qkv + (size_t)n * (3 * DIM);
  const float* q = row + h * DH;
  const float* k = q + DIM;
  const float* vv = q + 2 * DIM;
  float sq = 0.f, sk = 0.f;
  for (int j = 0; j < DH; ++j) { sq += q[j] * q[j]; sk += k[j] * k[j]; }
  float iq = 1.f / (sqrtf(sq) + 1e-12f);
  float ik = 1.f / (sqrtf(sk) + 1e-12f);
  size_t o = (size_t)n * DIM + h * DH;
  for (int j = 0; j < DH; ++j) {
    qn[o + j] = (bf16)(q[j] * iq);
    kn[o + j] = (bf16)(k[j] * ik);
    v[o + j] = vv[j];
  }
}

// ---------------- per-window C = sum qn kn^T, softmax -> A ----------------
__global__ void k_winC(const bf16* __restrict__ qn, const bf16* __restrict__ kn,
                       const int* __restrict__ offsets, const int* __restrict__ counts,
                       const float* __restrict__ temp, float* __restrict__ A) {
  int w = blockIdx.x;
  int tid = threadIdx.x;           // 128 threads: tid = h*16 + d
  int h = tid >> 4, d = tid & 15;
  int off = offsets[w], cnt = counts[w];
  float acc[16];
  for (int e = 0; e < 16; ++e) acc[e] = 0.f;
  for (int r = 0; r < cnt; ++r) {
    size_t n = (size_t)(off + r);
    float qv = (float)qn[n * DIM + h * DH + d];
    const bf16* kr = kn + n * DIM + h * DH;
    for (int e = 0; e < 16; ++e) acc[e] += qv * (float)kr[e];
  }
  float tp = temp[h];
  float mx = -3.0e38f;
  for (int e = 0; e < 16; ++e) { acc[e] *= tp; mx = fmaxf(mx, acc[e]); }
  float s = 0.f;
  for (int e = 0; e < 16; ++e) { acc[e] = expf(acc[e] - mx); s += acc[e]; }
  float inv = 1.f / s;
  float* out = A + ((size_t)w * NH + h) * 256 + d * 16;
  for (int e = 0; e < 16; ++e) out[e] = acc[e] * inv;
}

// ---------------- attn[n,h,d] = sum_e A[bwi[n],h,d,e] * v[n,h,e] ----------------
__global__ void k_attn(const float* __restrict__ A, const float* __restrict__ v,
                       const int* __restrict__ bwi, bf16* __restrict__ attn) {
  int i = blockIdx.x * blockDim.x + threadIdx.x;
  if (i >= NPTS * DIM) return;
  int n = i >> 7, hd = i & 127;
  int h = hd >> 4, d = hd & 15;
  int w = bwi[n];
  const float* Ar = A + ((size_t)w * NH + h) * 256 + d * 16;
  const float* vr = v + (size_t)n * DIM + h * DH;
  float acc = 0.f;
  for (int e = 0; e < 16; ++e) acc += Ar[e] * vr[e];
  attn[i] = (bf16)acc;
}

// ---------------- residual + layernorm (one 128-thread block per row) ----------------
__global__ void k_add_ln(const float* __restrict__ x, const float* __restrict__ y,
                         const float* __restrict__ g, const float* __restrict__ be,
                         float* __restrict__ outF, bf16* __restrict__ outB) {
  __shared__ float s1[128], s2[128];
  int n = blockIdx.x, t = threadIdx.x;
  size_t o = (size_t)n * DIM + t;
  float v = x[o] + (y ? y[o] : 0.f);
  s1[t] = v; s2[t] = v * v;
  __syncthreads();
  for (int st = 64; st > 0; st >>= 1) {
    if (t < st) { s1[t] += s1[t + st]; s2[t] += s2[t + st]; }
    __syncthreads();
  }
  float mean = s1[0] * (1.f / DIM);
  float var = s2[0] * (1.f / DIM) - mean * mean;
  float r = rsqrtf(var + EPS_);
  float out = (v - mean) * r * g[t] + be[t];
  if (outF) outF[o] = out;
  if (outB) outB[o] = (bf16)out;
}

// ---------------- sparse coord -> row map ----------------
__global__ void k_map_init(int* __restrict__ map) {
  int i = blockIdx.x * blockDim.x + threadIdx.x;
  if (i < MAPSZ) map[i] = -1;
}
__global__ void k_map_scatter(const int* __restrict__ coords, int* __restrict__ map) {
  int n = blockIdx.x * blockDim.x + threadIdx.x;
  if (n >= NPTS) return;
  int bi = coords[n * 3], yi = coords[n * 3 + 1], xi = coords[n * 3 + 2];
  map[(size_t)bi * GRID_ * GRID_ + yi * GRID_ + xi] = n;
}

// ---------------- depthwise sparse convs + layernorm -> bf16 ----------------
__global__ void k_conv_ln(const float* __restrict__ s2d, const int* __restrict__ coords,
                          const int* __restrict__ map,
                          const float* __restrict__ whw, const float* __restrict__ bhw,
                          const float* __restrict__ ww,  const float* __restrict__ bw,
                          const float* __restrict__ wh,  const float* __restrict__ bh,
                          const float* __restrict__ cg,  const float* __restrict__ cbe,
                          bf16* __restrict__ outB) {
  __shared__ float r1[128], r2[128];
  int n = blockIdx.x, c = threadIdx.x;
  int bi = coords[n * 3], yi = coords[n * 3 + 1], xi = coords[n * 3 + 2];
  const int* mp = map + (size_t)bi * GRID_ * GRID_;
  float val;
  if (c < GC) {                               // identity channels 0..31
    val = s2d[(size_t)n * DIM + c];
  } else if (c < 2 * GC) {                    // 3x3 on channels 32..63
    int cc = c - GC;
    float acc = bhw[cc];
    for (int ky = 0; ky < 3; ++ky)
      for (int kx = 0; kx < 3; ++kx) {
        int yy = yi + ky - 1, xx = xi + kx - 1;
        if (yy >= 0 && yy < GRID_ && xx >= 0 && xx < GRID_) {
          int nb = mp[yy * GRID_ + xx];
          if (nb >= 0) acc += whw[(ky * 3 + kx) * GC + cc] * s2d[(size_t)nb * DIM + GC + cc];
        }
      }
    val = acc;
  } else if (c < 3 * GC) {                    // 1xKS on channels 64..95
    int cc = c - 2 * GC;
    float acc = bw[cc];
    for (int kx = 0; kx < KS; ++kx) {
      int xx = xi + kx - PAD_;
      if (xx >= 0 && xx < GRID_) {
        int nb = mp[yi * GRID_ + xx];
        if (nb >= 0) acc += ww[kx * GC + cc] * s2d[(size_t)nb * DIM + 2 * GC + cc];
      }
    }
    val = acc;
  } else {                                    // KSx1 on channels 96..127
    int cc = c - 3 * GC;
    float acc = bh[cc];
    for (int ky = 0; ky < KS; ++ky) {
      int yy = yi + ky - PAD_;
      if (yy >= 0 && yy < GRID_) {
        int nb = mp[yy * GRID_ + xi];
        if (nb >= 0) acc += wh[ky * GC + cc] * s2d[(size_t)nb * DIM + 3 * GC + cc];
      }
    }
    val = acc;
  }
  r1[c] = val; r2[c] = val * val;
  __syncthreads();
  for (int st = 64; st > 0; st >>= 1) {
    if (c < st) { r1[c] += r1[c + st]; r2[c] += r2[c + st]; }
    __syncthreads();
  }
  float mean = r1[0] * (1.f / DIM);
  float var = r2[0] * (1.f / DIM) - mean * mean;
  float rr = rsqrtf(var + EPS_);
  outB[(size_t)n * DIM + c] = (bf16)((val - mean) * rr * cg[c] + cbe[c]);
}

// ---------------- host launcher ----------------
extern "C" void kernel_launch(void* const* d_in, const int* in_sizes, int n_in,
                              void* d_out, int out_size, void* d_ws, size_t ws_size,
                              hipStream_t stream) {
  const float* src      = (const float*)d_in[0];
  const int*   offsets  = (const int*)d_in[1];
  const int*   counts   = (const int*)d_in[2];
  const int*   bwi      = (const int*)d_in[3];
  const int*   win_pos  = (const int*)d_in[4];
  const int*   coords   = (const int*)d_in[5];
  // d_in[6]=batch_size, d_in[7]=block_id (unused)
  const float* pe    = (const float*)d_in[8];
  const float* w_qkv = (const float*)d_in[9];
  const float* b_qkv = (const float*)d_in[10];
  const float* temp  = (const float*)d_in[11];
  const float* w_proj= (const float*)d_in[12];
  const float* b_proj= (const float*)d_in[13];
  const float* g0    = (const float*)d_in[14];
  const float* be0   = (const float*)d_in[15];
  const float* w1    = (const float*)d_in[16];
  const float* b1    = (const float*)d_in[17];
  const float* w2    = (const float*)d_in[18];
  const float* b2    = (const float*)d_in[19];
  const float* g1    = (const float*)d_in[20];
  const float* be1   = (const float*)d_in[21];
  const float* chw_w = (const float*)d_in[22];
  const float* chw_b = (const float*)d_in[23];
  const float* cw_w  = (const float*)d_in[24];
  const float* cw_b  = (const float*)d_in[25];
  const float* ch_w  = (const float*)d_in[26];
  const float* ch_b  = (const float*)d_in[27];
  const float* cg    = (const float*)d_in[28];
  const float* cbe   = (const float*)d_in[29];
  const float* cw1   = (const float*)d_in[30];
  const float* cb1   = (const float*)d_in[31];
  const float* cw2   = (const float*)d_in[32];
  const float* cb2   = (const float*)d_in[33];
  const float* g2    = (const float*)d_in[34];
  const float* be2   = (const float*)d_in[35];

  char* ws = (char*)d_ws;
  bf16* BF0 = (bf16*)(ws + OFF_BF0);
  bf16* BF1 = (bf16*)(ws + OFF_BF1);
  float* QKV = (float*)(ws + OFF_QKV);
  float* AMAT = (float*)(ws + OFF_QKV);   // reuse after qknorm
  float* Y   = (float*)(ws + OFF_Y);
  bf16* QN   = (bf16*)(ws + OFF_QN);
  bf16* KN   = (bf16*)(ws + OFF_KN);
  float* V   = (float*)(ws + OFF_V);
  float* S1  = (float*)(ws + OFF_S1);
  float* S2  = (float*)(ws + OFF_S2);
  int*  MAP  = (int*)(ws + OFF_MAP);
  bf16* PW   = (bf16*)(ws + OFF_PW);

  const int TB = 256;
  auto cdiv = [](int a, int b) { return (a + b - 1) / b; };
  // waves per block = TB/32 = 8; tiles-per-wave grouping of 8 column tiles
  auto gemm_blocks = [&](int M) {
    int groups = cdiv(M / 16, 8);
    return cdiv((NPTS / 16) * groups, 8);
  };

  // 1) pack all weights to WMMA B-operand layout
  k_pack_w<<<cdiv(128 * 384, TB), TB, 0, stream>>>(w_qkv, PW + PW_QKV, 128, 384);
  k_pack_w<<<cdiv(128 * 128, TB), TB, 0, stream>>>(w_proj, PW + PW_PROJ, 128, 128);
  k_pack_w<<<cdiv(128 * 256, TB), TB, 0, stream>>>(w1, PW + PW_W1, 128, 256);
  k_pack_w<<<cdiv(256 * 128, TB), TB, 0, stream>>>(w2, PW + PW_W2, 256, 128);
  k_pack_w<<<cdiv(128 * 256, TB), TB, 0, stream>>>(cw1, PW + PW_CW1, 128, 256);
  k_pack_w<<<cdiv(256 * 128, TB), TB, 0, stream>>>(cw2, PW + PW_CW2, 256, 128);

  // 2) x = src + pe (bf16)
  k_embed<<<cdiv(NPTS * DIM, TB), TB, 0, stream>>>(src, win_pos, pe, BF0);

  // 3) qkv GEMM: [N,128]x[128,384]
  k_gemm_wmma<0, 4><<<gemm_blocks(384), TB, 0, stream>>>(BF0, PW + PW_QKV, b_qkv,
                                                         QKV, nullptr, NPTS, 384);
  // 4) split + normalize q,k per head
  k_qknorm<<<cdiv(NPTS * NH, TB), TB, 0, stream>>>(QKV, QN, KN, V);
  // 5) per-window outer-product sums + softmax
  k_winC<<<NWIN, 128, 0, stream>>>(QN, KN, offsets, counts, temp, AMAT);
  // 6) attn gather
  k_attn<<<cdiv(NPTS * DIM, TB), TB, 0, stream>>>(AMAT, V, bwi, BF0);
  // 7) proj GEMM [N,128]x[128,128]
  k_gemm_wmma<0, 4><<<gemm_blocks(128), TB, 0, stream>>>(BF0, PW + PW_PROJ, b_proj,
                                                         Y, nullptr, NPTS, 128);
  // 8) src1 = LN(src + proj)
  k_add_ln<<<NPTS, 128, 0, stream>>>(src, Y, g0, be0, S1, BF0);
  // 9) FFN1: relu(src1 @ w1 + b1) -> bf16
  k_gemm_wmma<1, 4><<<gemm_blocks(256), TB, 0, stream>>>(BF0, PW + PW_W1, b1,
                                                         nullptr, BF1, NPTS, 256);
  // 10) FFN1b: h1 @ w2 + b2 -> f32
  k_gemm_wmma<0, 8><<<gemm_blocks(128), TB, 0, stream>>>(BF1, PW + PW_W2, b2,
                                                         Y, nullptr, NPTS, 128);
  // 11) src2 = LN(src1 + ffn)
  k_add_ln<<<NPTS, 128, 0, stream>>>(S1, Y, g1, be1, S2, nullptr);
  // 12) sparse coord map
  k_map_init<<<cdiv(MAPSZ, TB), TB, 0, stream>>>(MAP);
  k_map_scatter<<<cdiv(NPTS, TB), TB, 0, stream>>>(coords, MAP);
  // 13) depthwise convs + LN -> bf16
  k_conv_ln<<<NPTS, 128, 0, stream>>>(S2, coords, MAP, chw_w, chw_b, cw_w, cw_b,
                                      ch_w, ch_b, cg, cbe, BF0);
  // 14) FFN2: relu(h @ cw1 + cb1) -> bf16
  k_gemm_wmma<1, 4><<<gemm_blocks(256), TB, 0, stream>>>(BF0, PW + PW_CW1, cb1,
                                                         nullptr, BF1, NPTS, 256);
  // 15) FFN2b: t1 @ cw2 + cb2 -> f32
  k_gemm_wmma<0, 8><<<gemm_blocks(128), TB, 0, stream>>>(BF1, PW + PW_CW2, cb2,
                                                         Y, nullptr, NPTS, 128);
  // 16) out = LN(src2 + ffn2)
  k_add_ln<<<NPTS, 128, 0, stream>>>(S2, Y, g2, be2, (float*)d_out, nullptr);
}
